// MultiHeadAttention_41712722379126
// MI455X (gfx1250) — matrix operand order repaired and checked by reading, not compiled, not run on hardware
//
#include <hip/hip_runtime.h>
#include <hip/hip_bf16.h>

// ---------------------------------------------------------------------------
// MHA + LayerNorm for MI455X (gfx1250, wave32, WMMA + TDM).
// B=4, L=2048, M=1024, H=16, D=64.
// ---------------------------------------------------------------------------

typedef __attribute__((ext_vector_type(16))) __bf16 v16bf;
typedef __attribute__((ext_vector_type(8)))  float  v8f;

struct alignas(16) U128 { unsigned int x[4]; };

union FragU {
  v16bf v;
  U128  q[2];
};

__device__ inline v16bf load_frag_2x16B(const __bf16* lo, const __bf16* hi) {
  FragU f;
  f.q[0] = *(const U128*)lo;
  f.q[1] = *(const U128*)hi;
  return f.v;
}

__device__ inline v16bf load_frag_32B(const __bf16* p) {
  FragU f;
  const U128* u = (const U128*)p;
  f.q[0] = u[0];
  f.q[1] = u[1];
  return f.v;
}

__device__ inline v8f wmma_bf16(v16bf a, v16bf b, v8f c) {
  // D = A(16x32 bf16) * B(32x16 bf16) + C(16x16 f32)
  return __builtin_amdgcn_wmma_f32_16x16x32_bf16(
      /*neg_a=*/false, a, /*neg_b=*/false, b,
      /*c_mod=*/(short)0, c, /*reuse_a=*/false, /*reuse_b=*/false);
}

#define LDSTR 48   // LDS row stride in bf16 elements (96B -> 16B-aligned rows)

// ---------------------------------------------------------------------------
// Tensor Data Mover (async tensor load to LDS). The D# pad feature is set to
// pad_interval = 16 DWORDs (64B = one 32-elem bf16 tile row) and
// pad_amount = 8 DWORDs (32B), reproducing the 96B (LDSTR=48) LDS row stride.
// ---------------------------------------------------------------------------
#if defined(__HIP_DEVICE_COMPILE__) && __has_builtin(__builtin_amdgcn_tensor_load_to_lds)
#define USE_TDM 1
#else
#define USE_TDM 0
#endif

#if USE_TDM
typedef unsigned int u32;
typedef __attribute__((ext_vector_type(4))) u32 u32x4;
typedef __attribute__((ext_vector_type(4))) int i32x4;
typedef __attribute__((ext_vector_type(8))) int i32x8;

__device__ inline void tdm_load_tile_bf16(
    u32 lds_byte_off, const void* gptr,
    u32 tile_d0, u32 tile_d1,        // tile extents (elements): d0 contiguous
    u32 tensor_d0, u32 tensor_d1,    // tensor extents (elements)
    u32 stride0)                     // elements between dim-1 lines
{
  unsigned long long ga = (unsigned long long)gptr;

  u32x4 g0;
  g0[0] = 1u;                                   // count=1 (valid user D#)
  g0[1] = lds_byte_off;                         // LDS destination (bytes)
  g0[2] = (u32)(ga & 0xFFFFFFFFu);              // global_addr[31:0]
  g0[3] = (u32)((ga >> 32) & 0x01FFFFFFu)       // global_addr[56:32]
        | (2u << 30);                           // type = 2 ("image")

  i32x8 g1;
  g1[0] = (int)((1u << 16)                      // data_size = 2 bytes
              | (1u << 20)                      // pad_enable
              | (3u << 22)                      // pad_interval: 16 DWORDs
              | (7u << 25));                    // pad_amount:   8 DWORDs
  g1[1] = (int)((tensor_d0 & 0xFFFFu) << 16);                    // dim0[15:0]
  g1[2] = (int)(((tensor_d0 >> 16) & 0xFFFFu)                    // dim0[31:16]
              | ((tensor_d1 & 0xFFFFu) << 16));                  // dim1[15:0]
  g1[3] = (int)(((tensor_d1 >> 16) & 0xFFFFu)                    // dim1[31:16]
              | ((tile_d0 & 0xFFFFu) << 16));                    // tile_dim0
  g1[4] = (int)(tile_d1 & 0xFFFFu);                              // tile_dim1
  g1[5] = (int)stride0;                                          // dim0 stride lo
  g1[6] = 0;                                                     // stride hi / dim1 stride
  g1[7] = 0;

  i32x4 z4 = {0, 0, 0, 0};
#if defined(__clang_major__) && (__clang_major__ >= 23)
  i32x8 z8 = {0, 0, 0, 0, 0, 0, 0, 0};
  __builtin_amdgcn_tensor_load_to_lds(g0, g1, z4, z4, z8, 0);
#else
  __builtin_amdgcn_tensor_load_to_lds(g0, g1, z4, z4, 0);
#endif
}
#endif  // USE_TDM

// ---------------------------------------------------------------------------
// Kernel 1: cast the four 1024x1024 weight matrices fp32 -> bf16
// ---------------------------------------------------------------------------
__global__ __launch_bounds__(256) void cast_weights(
    const float* wq, const float* wk, const float* wv, const float* wfc,
    __bf16* oq, __bf16* ok, __bf16* ov, __bf16* ofc)
{
  size_t i = (size_t)blockIdx.x * blockDim.x + threadIdx.x;  // 0 .. 4M-1
  size_t m = i >> 20;
  size_t e = i & ((1u << 20) - 1);
  const float* src = (m == 0) ? wq : (m == 1) ? wk : (m == 2) ? wv : wfc;
  __bf16*      dst = (m == 0) ? oq : (m == 1) ? ok : (m == 2) ? ov : ofc;
  dst[e] = (__bf16)src[e];
}

// ---------------------------------------------------------------------------
// Kernel 2: projection GEMM  Y[8192,1024] = X[8192,1024] @ W[1024,1024]
// blockDim=128 (4 waves). 128x128 block tile, 64x64 per wave (4x4 WMMA tiles).
// z = 0: q->QH, 1: k->KH, 2: v->VT (transposed store [b][h][d][L]).
// ---------------------------------------------------------------------------
__global__ __launch_bounds__(128) void proj_gemm(
    const float* qin, const float* kin, const float* vin,
    const __bf16* wq, const __bf16* wk, const __bf16* wv,
    __bf16* qh, __bf16* kh, __bf16* vt)
{
  __shared__ __align__(16) __bf16 As[128 * LDSTR];
  __shared__ __align__(16) __bf16 Bs[128 * LDSTR];

  const int z = blockIdx.z;
  const float*  X = (z == 0) ? qin : (z == 1) ? kin : vin;
  const __bf16* W = (z == 0) ? wq  : (z == 1) ? wk  : wv;

  const int t    = threadIdx.x;
  const int wave = t >> 5, lane = t & 31;
  const int half = lane >> 4, ln = lane & 15;
  const int wm   = wave >> 1, wn = wave & 1;
  const int mb   = blockIdx.x * 128, nb = blockIdx.y * 128;

  v8f acc[4][4];
#pragma unroll
  for (int i = 0; i < 4; ++i)
#pragma unroll
    for (int j = 0; j < 4; ++j) acc[i][j] = {};

  for (int kk = 0; kk < 1024; kk += 32) {
    __syncthreads();
#pragma unroll 4
    for (int i = 0; i < 32; ++i) {           // A: 128x32, fp32->bf16
      int e = i * 128 + t, r = e >> 5, c = e & 31;
      As[r * LDSTR + c] = (__bf16)X[(size_t)(mb + r) * 1024 + kk + c];
    }
#pragma unroll 4
    for (int i = 0; i < 32; ++i) {           // B: 32x128, stored transposed [n][k]
      Bs[t * LDSTR + i] = W[(size_t)(kk + i) * 1024 + nb + t];
    }
    if (kk + 32 < 1024) {
      __builtin_prefetch(&W[(size_t)(kk + 32) * 1024 + nb + t], 0, 0);
      __builtin_prefetch(&X[(size_t)(mb + (t >> 2)) * 1024 + kk + 32], 0, 0);
    }
    __syncthreads();

    v16bf a[4], b[4];
#pragma unroll
    for (int mt = 0; mt < 4; ++mt) {
      int m = wm * 64 + mt * 16 + ln;
      a[mt] = load_frag_2x16B(&As[m * LDSTR + half * 16],
                              &As[m * LDSTR + half * 16 + 8]);
    }
#pragma unroll
    for (int nt = 0; nt < 4; ++nt) {
      int n = wn * 64 + nt * 16 + ln;
      b[nt] = load_frag_32B(&Bs[n * LDSTR + half * 16]);
    }
#pragma unroll
    for (int mt = 0; mt < 4; ++mt)
#pragma unroll
      for (int nt = 0; nt < 4; ++nt)
        acc[mt][nt] = wmma_bf16(a[mt], b[nt], acc[mt][nt]);
  }

#pragma unroll
  for (int mt = 0; mt < 4; ++mt)
#pragma unroll
    for (int nt = 0; nt < 4; ++nt)
#pragma unroll
      for (int r = 0; r < 8; ++r) {
        int row = mb + wm * 64 + mt * 16 + r + 8 * half;
        int col = nb + wn * 64 + nt * 16 + ln;
        __bf16 val = (__bf16)acc[mt][nt][r];
        if (z == 2) {   // V stored transposed: [b][h][d][L]
          int bi = row >> 11, l = row & 2047;
          int hh = col >> 6,  d = col & 63;
          vt[((((size_t)bi * 16 + hh) * 64 + d) << 11) + l] = val;
        } else {
          __bf16* Y = (z == 0) ? qh : kh;
          Y[(size_t)row * 1024 + col] = val;
        }
      }
}

// ---------------------------------------------------------------------------
// Kernel 3: flash attention. One wave per (b, h, 16-row q block).
// 32-key chunks: S = Q@K^T (4 WMMA), online softmax, O += P@V (4 WMMA).
// ---------------------------------------------------------------------------
__global__ __launch_bounds__(128) void attention(
    const __bf16* qh, const __bf16* kh, const __bf16* vt, __bf16* ao)
{
  __shared__ __align__(16) __bf16 Pl[4][16 * LDSTR];  // per-wave P scratch

  const int t = threadIdx.x;
  const int wave = t >> 5, lane = t & 31;
  const int half = lane >> 4, ln = lane & 15;

  const int task = blockIdx.x * 4 + wave;   // 0..8191
  const int qb = task & 127;
  const int h  = (task >> 7) & 15;
  const int b  = task >> 11;
  const size_t qrow = (size_t)b * 2048 + qb * 16;

  // Q fragments for kdim [0,32) and [32,64)
  v16bf aq[2];
  {
    const __bf16* qbase = qh + (qrow + ln) * 1024 + h * 64;
    aq[0] = load_frag_2x16B(qbase + half * 16,      qbase + 8 + half * 16);
    aq[1] = load_frag_2x16B(qbase + 32 + half * 16, qbase + 40 + half * 16);
  }

  v8f vout[4];
#pragma unroll
  for (int i = 0; i < 4; ++i) vout[i] = {};
  float rowmax[8], rowsum[8];
#pragma unroll
  for (int r = 0; r < 8; ++r) { rowmax[r] = -1e30f; rowsum[r] = 0.0f; }

  __bf16* P = &Pl[wave][0];
  const __bf16* vbase =
      vt + ((((size_t)b * 16 + h) * 64) << 11);   // [d][L] plane for this (b,h)

  for (int kc = 0; kc < 64; ++kc) {
    // ---- scores S[16 x 32] as two 16x16 accumulators ----
    v8f s[2];
#pragma unroll
    for (int nt = 0; nt < 2; ++nt) {
      size_t keyrow = (size_t)b * 2048 + kc * 32 + nt * 16 + ln;
      const __bf16* kbase = kh + keyrow * 1024 + h * 64;
      v16bf bk0 = load_frag_32B(kbase + half * 16);        // kdim [0,32)
      v16bf bk1 = load_frag_32B(kbase + 32 + half * 16);   // kdim [32,64)
      v8f c = {};
      c = wmma_bf16(aq[0], bk0, c);
      c = wmma_bf16(aq[1], bk1, c);
      s[nt] = c * 0.125f;   // 1/sqrt(64)
    }

    // ---- online softmax: element r <-> row (r + 8*half); cols across 16 lanes
    float p0[8], p1[8];
#pragma unroll
    for (int r = 0; r < 8; ++r) {
      float mx = fmaxf(s[0][r], s[1][r]);
#pragma unroll
      for (int off = 1; off < 16; off <<= 1)
        mx = fmaxf(mx, __shfl_xor(mx, off, 32));
      float mnew  = fmaxf(rowmax[r], mx);
      float alpha = __expf(rowmax[r] - mnew);
      rowmax[r] = mnew;
      p0[r] = __expf(s[0][r] - mnew);
      p1[r] = __expf(s[1][r] - mnew);
      float ps = p0[r] + p1[r];
#pragma unroll
      for (int off = 1; off < 16; off <<= 1)
        ps += __shfl_xor(ps, off, 32);
      rowsum[r] = rowsum[r] * alpha + ps;
#pragma unroll
      for (int nt2 = 0; nt2 < 4; ++nt2) vout[nt2][r] *= alpha;
    }

    // ---- relayout P (C-layout) -> A-layout through wave-local LDS ----
#pragma unroll
    for (int r = 0; r < 8; ++r) {
      int m = r + 8 * half;
      P[m * LDSTR + ln]      = (__bf16)p0[r];
      P[m * LDSTR + 16 + ln] = (__bf16)p1[r];
    }
    // same-wave DS ops are in-order; compiler inserts s_wait_dscnt for the RAW
    v16bf ap = load_frag_2x16B(&P[ln * LDSTR + half * 16],
                               &P[ln * LDSTR + half * 16 + 8]);

    // ---- O += P @ V  (V^T layout makes B-frags contiguous along L) ----
    const __bf16* vchunk = vbase + kc * 32 + half * 16;
#pragma unroll
    for (int nt2 = 0; nt2 < 4; ++nt2) {
      const __bf16* vp = vchunk + ((size_t)(nt2 * 16 + ln) << 11);
      v16bf bv = load_frag_32B(vp);
      vout[nt2] = wmma_bf16(ap, bv, vout[nt2]);
    }
  }

  // ---- normalize and store attention output [rows, H*64] bf16 ----
#pragma unroll
  for (int nt2 = 0; nt2 < 4; ++nt2)
#pragma unroll
    for (int r = 0; r < 8; ++r) {
      int m = r + 8 * half;
      float o = vout[nt2][r] / rowsum[r];
      ao[(qrow + m) * 1024 + h * 64 + nt2 * 16 + ln] = (__bf16)o;
    }
}

// ---------------------------------------------------------------------------
// Kernel 4: out = AO @ fc_w + fc_b + residual(q), fp32 into d_out.
// A tile (128x32 bf16) is DMA'd into LDS by the Tensor Data Mover (wave 0
// issues the D#), overlapped with the cooperative B-tile fill; synced with
// s_wait_tensorcnt + workgroup barrier. Dynamic LDS so As sits at offset 0.
// ---------------------------------------------------------------------------
__global__ __launch_bounds__(128) void fc_residual(
    const __bf16* ao, const __bf16* wfc, const float* fcb,
    const float* res, float* out)
{
  extern __shared__ __bf16 ldsbuf[];
  __bf16* As = ldsbuf;                 // LDS bytes [0, 12288)
  __bf16* Bs = ldsbuf + 128 * LDSTR;   // LDS bytes [12288, 24576)

  const int t    = threadIdx.x;
  const int wave = t >> 5, lane = t & 31;
  const int half = lane >> 4, ln = lane & 15;
  const int wm   = wave >> 1, wn = wave & 1;
  const int mb   = blockIdx.x * 128, nb = blockIdx.y * 128;

  v8f acc[4][4];
#pragma unroll
  for (int i = 0; i < 4; ++i)
#pragma unroll
    for (int j = 0; j < 4; ++j) acc[i][j] = {};

  for (int kk = 0; kk < 1024; kk += 32) {
    __syncthreads();
#if USE_TDM
    if (wave == 0) {
      // async DMA: 128 rows x 32 cols bf16 from AO[mb.., kk..] into As
      tdm_load_tile_bf16(/*lds_byte_off=*/0u,
                         ao + (size_t)mb * 1024 + kk,
                         /*tile_d0=*/32u, /*tile_d1=*/128u,
                         /*tensor_d0=*/1024u, /*tensor_d1=*/8192u,
                         /*stride0=*/1024u);
    }
#else
#pragma unroll 4
    for (int i = 0; i < 32; ++i) {
      int e = i * 128 + t, r = e >> 5, c = e & 31;
      As[r * LDSTR + c] = ao[(size_t)(mb + r) * 1024 + kk + c];
    }
#endif
#pragma unroll 4
    for (int i = 0; i < 32; ++i) {       // B: 32x128 -> LDS transposed [n][k]
      Bs[t * LDSTR + i] = wfc[(size_t)(kk + i) * 1024 + nb + t];
    }
    if (kk + 32 < 1024)
      __builtin_prefetch(&wfc[(size_t)(kk + 32) * 1024 + nb + t], 0, 0);
#if USE_TDM
    if (wave == 0) __builtin_amdgcn_s_wait_tensorcnt(0);
#endif
    __syncthreads();

    v16bf a[4], b[4];
#pragma unroll
    for (int mt = 0; mt < 4; ++mt) {
      int m = wm * 64 + mt * 16 + ln;
      a[mt] = load_frag_2x16B(&As[m * LDSTR + half * 16],
                              &As[m * LDSTR + half * 16 + 8]);
    }
#pragma unroll
    for (int nt = 0; nt < 4; ++nt) {
      int n = wn * 64 + nt * 16 + ln;
      b[nt] = load_frag_32B(&Bs[n * LDSTR + half * 16]);
    }
#pragma unroll
    for (int mt = 0; mt < 4; ++mt)
#pragma unroll
      for (int nt = 0; nt < 4; ++nt)
        acc[mt][nt] = wmma_bf16(a[mt], b[nt], acc[mt][nt]);
  }

#pragma unroll
  for (int mt = 0; mt < 4; ++mt)
#pragma unroll
    for (int nt = 0; nt < 4; ++nt)
#pragma unroll
      for (int r = 0; r < 8; ++r) {
        int row = mb + wm * 64 + mt * 16 + r + 8 * half;
        int col = nb + wn * 64 + nt * 16 + ln;
        float v = acc[mt][nt][r] + fcb[col] + res[(size_t)row * 1024 + col];
        out[(size_t)row * 1024 + col] = v;
      }
}

// ---------------------------------------------------------------------------
// Kernel 5: LayerNorm in place over the last dim (1024) of d_out
// ---------------------------------------------------------------------------
__global__ __launch_bounds__(256) void layernorm_inplace(
    float* x, const float* g, const float* bb)
{
  __shared__ float red[256];
  const int row = blockIdx.x;
  float* xr = x + (size_t)row * 1024;

  float v[4];
  float s = 0.0f;
#pragma unroll
  for (int i = 0; i < 4; ++i) { v[i] = xr[threadIdx.x + i * 256]; s += v[i]; }

  red[threadIdx.x] = s;
  __syncthreads();
  for (int st = 128; st > 0; st >>= 1) {
    if (threadIdx.x < st) red[threadIdx.x] += red[threadIdx.x + st];
    __syncthreads();
  }
  float mu = red[0] * (1.0f / 1024.0f);
  __syncthreads();

  float s2 = 0.0f;
#pragma unroll
  for (int i = 0; i < 4; ++i) { float d = v[i] - mu; s2 += d * d; }
  red[threadIdx.x] = s2;
  __syncthreads();
  for (int st = 128; st > 0; st >>= 1) {
    if (threadIdx.x < st) red[threadIdx.x] += red[threadIdx.x + st];
    __syncthreads();
  }
  float rs = rsqrtf(red[0] * (1.0f / 1024.0f) + 1e-5f);

#pragma unroll
  for (int i = 0; i < 4; ++i) {
    int c = threadIdx.x + i * 256;
    xr[c] = (v[i] - mu) * rs * g[c] + bb[c];
  }
}

// ---------------------------------------------------------------------------
// Host launcher
// ---------------------------------------------------------------------------
extern "C" void kernel_launch(void* const* d_in, const int* in_sizes, int n_in,
                              void* d_out, int out_size, void* d_ws, size_t ws_size,
                              hipStream_t stream) {
  const float* q    = (const float*)d_in[0];
  const float* k    = (const float*)d_in[1];
  const float* v    = (const float*)d_in[2];
  const float* W_qs = (const float*)d_in[3];
  const float* W_ks = (const float*)d_in[4];
  const float* W_vs = (const float*)d_in[5];
  const float* fc_w = (const float*)d_in[6];
  const float* fc_b = (const float*)d_in[7];
  const float* ln_g = (const float*)d_in[8];
  const float* ln_b = (const float*)d_in[9];
  float* out = (float*)d_out;

  char* ws = (char*)d_ws;
  const size_t MB = 1024 * 1024;
  __bf16* WQ  = (__bf16*)(ws + 0 * MB);
  __bf16* WK  = (__bf16*)(ws + 2 * MB);
  __bf16* WV  = (__bf16*)(ws + 4 * MB);
  __bf16* WFC = (__bf16*)(ws + 6 * MB);
  __bf16* QH  = (__bf16*)(ws + 8 * MB);    // [8192, 1024] bf16 (16 MB)
  __bf16* KH  = (__bf16*)(ws + 24 * MB);   // [8192, 1024] bf16
  __bf16* VT  = (__bf16*)(ws + 40 * MB);   // [B][H][64][2048] bf16
  __bf16* AO  = (__bf16*)(ws + 56 * MB);   // [8192, 1024] bf16

  cast_weights<<<16384, 256, 0, stream>>>(W_qs, W_ks, W_vs, fc_w,
                                          WQ, WK, WV, WFC);

  proj_gemm<<<dim3(64, 8, 3), 128, 0, stream>>>(q, k, v, WQ, WK, WV,
                                                QH, KH, VT);

  attention<<<2048, 128, 0, stream>>>(QH, KH, VT, AO);

  fc_residual<<<dim3(64, 8), 128, 24576, stream>>>(AO, WFC, fc_b, q, out);

  layernorm_inplace<<<8192, 256, 0, stream>>>(out, ln_g, ln_b);
}
